// IPAdapterAttnProcessor_19164144074757
// MI455X (gfx1250) — matrix operand order repaired and checked by reading, not compiled, not run on hardware
//
#include <hip/hip_runtime.h>

typedef unsigned short u16;
typedef __attribute__((ext_vector_type(16))) __bf16 v16bf;
typedef __attribute__((ext_vector_type(8)))  float  v8f;

union Frag {
  v16bf v;
  uint4 q[2];
};

__device__ __forceinline__ u16 f2bf(float f) {
  union { float f; unsigned u; } v; v.f = f;
  unsigned r = v.u + 0x7FFFu + ((v.u >> 16) & 1u);
  return (u16)(r >> 16);
}

__device__ __forceinline__ v8f wmma_bf16(const Frag& a, const Frag& b, v8f c) {
  return __builtin_amdgcn_wmma_f32_16x16x32_bf16(false, a.v, false, b.v,
                                                 (short)0, c, false, false);
}

// ---------------------------------------------------------------------------
// Tensor Data Mover staging (gfx1250). Guarded: falls back to per-lane
// loads + ds_store if the toolchain lacks the TDM builtins.
// ---------------------------------------------------------------------------
#if defined(__AMDGCN__) && __has_builtin(__builtin_amdgcn_tensor_load_to_lds) && \
    __has_builtin(__builtin_amdgcn_s_wait_tensorcnt)
#define USE_TDM 1
#else
#define USE_TDM 0
#endif

#if USE_TDM
typedef __attribute__((ext_vector_type(4))) unsigned vu4;
typedef __attribute__((ext_vector_type(4))) int vi4;
typedef __attribute__((ext_vector_type(8))) int vi8;

__device__ __forceinline__ unsigned lds_off(const void* p) {
  // generic LDS pointer: low 32 bits hold the LDS byte offset
  return (unsigned)(size_t)p;
}

// DMA a [128 rows x 32 cols] bf16 tile (row stride `ld` elements) into LDS,
// padding 4 DWORDs after every 16 DWORDs (-> 40-halfword LDS row stride).
// Rows/cols beyond (rows_rem, cols_rem) are zero-filled by TDM OOB handling.
__device__ __forceinline__ void tdm_load_tile(const u16* gsrc, unsigned ldsaddr,
                                              int cols_rem, int rows_rem, int ld) {
  union { unsigned w[4]; vu4 v; } g0;
  union { unsigned w[8]; vi8 v; } g1;
  union { unsigned w[4]; vi4 v; } g23;
  unsigned long long ga = (unsigned long long)(size_t)gsrc;
  g0.w[0] = 1u;                                   // count=1 (valid, user mode)
  g0.w[1] = ldsaddr;                              // lds_addr
  g0.w[2] = (unsigned)ga;                         // global_addr[31:0]
  g0.w[3] = (unsigned)(ga >> 32) | (2u << 30);    // global_addr[56:32] | type=2
  // data_size=2B, pad_enable, pad_interval: 16 DWORDs, pad_amount: 4 DWORDs
  g1.w[0] = (1u << 16) | (1u << 20) | (3u << 22) | (3u << 25);
  g1.w[1] = ((unsigned)cols_rem & 0xFFFFu) << 16;             // tensor_dim0 lo
  g1.w[2] = (((unsigned)cols_rem >> 16) & 0xFFFFu) |
            (((unsigned)rows_rem & 0xFFFFu) << 16);           // dim0 hi | dim1 lo
  g1.w[3] = (((unsigned)rows_rem >> 16) & 0xFFFFu) |
            (32u << 16);                                      // dim1 hi | tile_dim0=32
  g1.w[4] = 128u;                                             // tile_dim1=128
  g1.w[5] = (unsigned)ld;                                     // tensor_dim0_stride lo
  g1.w[6] = 0u;                                               // stride hi | dim1_stride lo
  g1.w[7] = 0u;
  g23.w[0] = 0; g23.w[1] = 0; g23.w[2] = 0; g23.w[3] = 0;
#if __clang_major__ >= 23
  union { unsigned w[8]; vi8 v; } gx = {};
  __builtin_amdgcn_tensor_load_to_lds(g0.v, g1.v, g23.v, g23.v, gx.v, 0);
#else
  __builtin_amdgcn_tensor_load_to_lds(g0.v, g1.v, g23.v, g23.v, 0);
#endif
}
#endif  // USE_TDM

// ---------------------------------------------------------------------------
// C[M,N] (f32 or bf16) = A_bf16[M,K] @ Bt_bf16[N,K]^T (+ bias)
// 256 threads = 8 waves; block tile 128x128, BK=32; wave tile 32x64.
// TDM path: double-buffered LDS, wave 0 drives the DMA, TENSORcnt sync.
// ---------------------------------------------------------------------------
template<bool OUT_BF16, bool HAS_BIAS>
__global__ __launch_bounds__(256)
void gemm_wmma(const u16* __restrict__ A, const u16* __restrict__ Bt,
               void* __restrict__ Cout, const float* __restrict__ bias,
               int M, int N, int K) {
#if USE_TDM
  __shared__ u16 sA[2][128 * 40];   // [m][k], padded row stride 40 halves
  __shared__ u16 sB[2][128 * 40];   // [n][k]
#else
  __shared__ u16 sA[1][128 * 40];
  __shared__ u16 sB[1][128 * 40];
#endif
  const int bm = blockIdx.x * 128, bn = blockIdx.y * 128;
  const int tid = threadIdx.x, lane = tid & 31, wave = tid >> 5;
  const int wm = (wave & 3) * 32;     // 4 waves along M
  const int wn = (wave >> 2) * 64;    // 2 waves along N
  const int lm = lane & 15, lh = lane >> 4;
  const int kbA = (lane < 16) ? 0 : 8;    // A-layout K chunk base
  const int kbB = (lane < 16) ? 0 : 16;   // unused; B frags also load at kbA
  v8f zero = {};
  v8f acc[2][4];
#pragma unroll
  for (int i = 0; i < 2; ++i)
#pragma unroll
    for (int j = 0; j < 4; ++j) acc[i][j] = zero;

  int cur = 0;
#if USE_TDM
  if (wave == 0) {
    tdm_load_tile(A + (size_t)bm * K, lds_off(&sA[0][0]), K, M - bm, K);
    tdm_load_tile(Bt + (size_t)bn * K, lds_off(&sB[0][0]), K, N - bn, K);
  }
#endif

  for (int k0 = 0; k0 < K; k0 += 32) {
#if USE_TDM
    const bool more = (k0 + 32) < K;
    if (wave == 0) {
      if (more) {   // kick off next tile's DMA before waiting on current
        tdm_load_tile(A + (size_t)bm * K + k0 + 32, lds_off(&sA[cur ^ 1][0]),
                      K - k0 - 32, M - bm, K);
        tdm_load_tile(Bt + (size_t)bn * K + k0 + 32, lds_off(&sB[cur ^ 1][0]),
                      K - k0 - 32, N - bn, K);
        __builtin_amdgcn_s_wait_tensorcnt(2);   // current tile's 2 DMAs done
      } else {
        __builtin_amdgcn_s_wait_tensorcnt(0);
      }
    }
    __syncthreads();
#else
    __syncthreads();
#pragma unroll
    for (int i = 0; i < 2; ++i) {
      int idx = tid + i * 256;                 // 512 chunks of 8 bf16
      int r = idx >> 2, c = (idx & 3) * 8;
      uint4 va = make_uint4(0, 0, 0, 0), vb = make_uint4(0, 0, 0, 0);
      if (bm + r < M) va = *(const uint4*)(A + (size_t)(bm + r) * K + k0 + c);
      if (bn + r < N) vb = *(const uint4*)(Bt + (size_t)(bn + r) * K + k0 + c);
      *(uint4*)(&sA[0][r * 40 + c]) = va;
      *(uint4*)(&sB[0][r * 40 + c]) = vb;
    }
    if (k0 + 32 < K) {                         // global_prefetch next K tile
      int r = tid >> 1;
      if (bm + r < M) __builtin_prefetch(A + (size_t)(bm + r) * K + k0 + 32, 0, 0);
      if (bn + r < N) __builtin_prefetch(Bt + (size_t)(bn + r) * K + k0 + 32, 0, 0);
    }
    __syncthreads();
#endif

    Frag fa[2], fb[4];
#pragma unroll
    for (int i = 0; i < 2; ++i) {
      const u16* p = &sA[cur][(wm + i * 16 + lm) * 40];
      fa[i].q[0] = *(const uint4*)(p + kbA);
      fa[i].q[1] = *(const uint4*)(p + kbA + 16);
    }
#pragma unroll
    for (int j = 0; j < 4; ++j) {
      const u16* p = &sB[cur][(wn + j * 16 + lm) * 40];
      fb[j].q[0] = *(const uint4*)(p + kbA);
      fb[j].q[1] = *(const uint4*)(p + kbA + 16);
    }
#pragma unroll
    for (int i = 0; i < 2; ++i)
#pragma unroll
      for (int j = 0; j < 4; ++j)
        acc[i][j] = wmma_bf16(fa[i], fb[j], acc[i][j]);

#if USE_TDM
    __syncthreads();    // all waves done with sX[cur] before it is re-filled
    cur ^= 1;
#endif
  }

  // C layout: vgpr e, lane -> row M = e + 8*(lane/16), col N = lane%16
#pragma unroll
  for (int i = 0; i < 2; ++i)
#pragma unroll
    for (int j = 0; j < 4; ++j) {
      int rbase = bm + wm + i * 16 + lh * 8;
      int cN = bn + wn + j * 16 + lm;
      float bval = HAS_BIAS ? bias[cN] : 0.f;
#pragma unroll
      for (int e = 0; e < 8; ++e) {
        int r = rbase + e;
        if (r < M) {
          float v = acc[i][j][e] + bval;
          if (OUT_BF16) ((u16*)Cout)[(size_t)r * N + cN] = f2bf(v);
          else          ((float*)Cout)[(size_t)r * N + cN] = v;
        }
      }
    }
}

// ---------------------------------------------------------------------------
// fp32 -> bf16 bulk convert (n divisible by 4)
// ---------------------------------------------------------------------------
__global__ __launch_bounds__(256)
void cvt_f32_bf16(const float* __restrict__ in, u16* __restrict__ out, long long n) {
  long long i = ((long long)blockIdx.x * 256 + threadIdx.x) * 4;
  if (i + 3 < n) {
    float4 v = *(const float4*)(in + i);
    union { u16 h[4]; uint2 u; } pk;
    pk.h[0] = f2bf(v.x); pk.h[1] = f2bf(v.y);
    pk.h[2] = f2bf(v.z); pk.h[3] = f2bf(v.w);
    *(uint2*)(out + i) = pk.u;
  }
}

// W[K][N] f32  ->  Wt[N][K] bf16 (K,N multiples of 32)
__global__ __launch_bounds__(256)
void transpose_cvt(const float* __restrict__ W, u16* __restrict__ Wt, int K, int N) {
  __shared__ float t[32][33];
  int n0 = blockIdx.x * 32, k0 = blockIdx.y * 32;
  int tx = threadIdx.x & 31, ty = threadIdx.x >> 5;  // ty: 0..7
#pragma unroll
  for (int i = 0; i < 32; i += 8)
    t[ty + i][tx] = W[(size_t)(k0 + ty + i) * N + n0 + tx];
  __syncthreads();
#pragma unroll
  for (int i = 0; i < 32; i += 8)
    Wt[(size_t)(n0 + ty + i) * K + k0 + tx] = f2bf(t[tx][ty + i]);
}

// ---------------------------------------------------------------------------
// Fused cross-attention: one block per (b, h, 64 q-rows); 4 waves x 16 rows.
// Scores + softmax + PV in WMMA; accO accumulates text + ip branches.
// ---------------------------------------------------------------------------
template<int T, int NT, int KT>     // keys, 16-col score tiles, 32-deep PV steps
__device__ __forceinline__ void attend_branch(
    const u16* __restrict__ Kg, const u16* __restrict__ Vg,
    int b, int h, int tid, int lane,
    u16* lK, u16* lV, u16* pw, const Frag* qf, v8f* accO) {
  const int lm = lane & 15, lh = lane >> 4;
  const int kbA = (lane < 16) ? 0 : 8;
  const int kbB = (lane < 16) ? 0 : 16;
  constexpr int TPAD = KT * 32;     // P/V padded depth (96 or 32)

  // stage K as [t][64] bf16 (zero pad t >= T)
  for (int idx = tid; idx < NT * 16 * 8; idx += 128) {
    int t = idx >> 3, c = (idx & 7) * 8;
    uint4 v = make_uint4(0, 0, 0, 0);
    if (t < T) v = *(const uint4*)(Kg + ((size_t)b * T + t) * 1280 + h * 64 + c);
    *(uint4*)(&lK[t * 64 + c]) = v;
  }
  // stage V transposed as [d][96] bf16 (zero pad t >= T)
  for (int idx = tid; idx < 64 * TPAD; idx += 128) {
    int d = idx / TPAD, t = idx - d * TPAD;
    u16 v = 0;
    if (t < T) v = Vg[((size_t)b * T + t) * 1280 + h * 64 + d];
    lV[d * 96 + t] = v;
  }
  __syncthreads();

  // scores: Q[16,64] @ K^T[64, NT*16]
  v8f zero = {};
  v8f accS[NT];
#pragma unroll
  for (int j = 0; j < NT; ++j) accS[j] = zero;
#pragma unroll
  for (int j = 0; j < NT; ++j) {
    const u16* kp = &lK[(j * 16 + lm) * 64];
#pragma unroll
    for (int kk = 0; kk < 2; ++kk) {
      Frag bk;
      bk.q[0] = *(const uint4*)(kp + kk * 32 + kbA);
      bk.q[1] = *(const uint4*)(kp + kk * 32 + kbA + 16);
      accS[j] = wmma_bf16(qf[kk], bk, accS[j]);
    }
  }

  // masked row softmax (rows live in 16-lane half-wave groups)
  const float scale = 0.125f;   // 1/sqrt(64)
#pragma unroll
  for (int j = 0; j < NT; ++j) {
    bool valid = (j * 16 + lm) < T;
#pragma unroll
    for (int e = 0; e < 8; ++e)
      accS[j][e] = valid ? accS[j][e] * scale : -1e30f;
  }
#pragma unroll
  for (int e = 0; e < 8; ++e) {
    float m = accS[0][e];
#pragma unroll
    for (int j = 1; j < NT; ++j) m = fmaxf(m, accS[j][e]);
#pragma unroll
    for (int o = 1; o < 16; o <<= 1) m = fmaxf(m, __shfl_xor(m, o, 32));
    float s = 0.f;
#pragma unroll
    for (int j = 0; j < NT; ++j) {
      float p = __expf(accS[j][e] - m);
      p = ((j * 16 + lm) < T) ? p : 0.f;
      accS[j][e] = p;
      s += p;
    }
#pragma unroll
    for (int o = 1; o < 16; o <<= 1) s += __shfl_xor(s, o, 32);
    float inv = 1.f / s;
#pragma unroll
    for (int j = 0; j < NT; ++j) accS[j][e] *= inv;
  }

  // P -> per-wave LDS [16][96] bf16 (zero the 16 padded cols), reload as A-frags
  for (int idx = lane; idx < 16 * 16; idx += 32) {
    int r = idx >> 4, c = NT * 16 + (idx & 15);
    pw[r * 96 + c] = 0;
  }
#pragma unroll
  for (int j = 0; j < NT; ++j)
#pragma unroll
    for (int e = 0; e < 8; ++e)
      pw[(e + 8 * lh) * 96 + j * 16 + lm] = f2bf(accS[j][e]);

  // out += P[16,TPAD] @ V[TPAD,64]
#pragma unroll
  for (int ks = 0; ks < KT; ++ks) {
    Frag pa;
    const u16* pp = &pw[lm * 96 + ks * 32 + kbA];
    pa.q[0] = *(const uint4*)(pp);
    pa.q[1] = *(const uint4*)(pp + 16);
#pragma unroll
    for (int n = 0; n < 4; ++n) {
      Frag bv;
      const u16* vp = &lV[(n * 16 + lm) * 96 + ks * 32 + kbA];
      bv.q[0] = *(const uint4*)(vp);
      bv.q[1] = *(const uint4*)(vp + 16);
      accO[n] = wmma_bf16(pa, bv, accO[n]);
    }
  }
}

__global__ __launch_bounds__(128)
void attn_wmma(const u16* __restrict__ Q, const u16* __restrict__ Ktxt,
               const u16* __restrict__ Vtxt, const u16* __restrict__ Kip,
               const u16* __restrict__ Vip, u16* __restrict__ Out) {
  __shared__ u16 lK[80 * 64];
  __shared__ u16 lV[64 * 96];
  __shared__ u16 lP[4 * 16 * 96];
  const int b = blockIdx.z, h = blockIdx.y;
  const int tid = threadIdx.x, lane = tid & 31, wave = tid >> 5;
  const int qrow = blockIdx.x * 64 + wave * 16;
  const int lm = lane & 15, lh = lane >> 4;
  const int kbA = (lane < 16) ? 0 : 8;

  Frag qf[2];
  {
    const u16* qp = Q + ((size_t)b * 4096 + qrow + lm) * 1280 + h * 64;
#pragma unroll
    for (int kk = 0; kk < 2; ++kk) {
      qf[kk].q[0] = *(const uint4*)(qp + kk * 32 + kbA);
      qf[kk].q[1] = *(const uint4*)(qp + kk * 32 + kbA + 16);
    }
  }
  v8f zero = {};
  v8f accO[4];
#pragma unroll
  for (int n = 0; n < 4; ++n) accO[n] = zero;
  u16* pw = &lP[wave * 16 * 96];

  attend_branch<77, 5, 3>(Ktxt, Vtxt, b, h, tid, lane, lK, lV, pw, qf, accO);
  __syncthreads();
  attend_branch<4, 1, 1>(Kip, Vip, b, h, tid, lane, lK, lV, pw, qf, accO);

  u16* op = Out + ((size_t)b * 4096 + qrow) * 1280 + h * 64;
#pragma unroll
  for (int n = 0; n < 4; ++n)
#pragma unroll
    for (int e = 0; e < 8; ++e)
      op[(size_t)(e + 8 * lh) * 1280 + n * 16 + lm] = f2bf(accO[n][e]);
}

// ---------------------------------------------------------------------------
extern "C" void kernel_launch(void* const* d_in, const int* in_sizes, int n_in,
                              void* d_out, int out_size, void* d_ws, size_t ws_size,
                              hipStream_t stream) {
  (void)in_sizes; (void)n_in; (void)out_size; (void)ws_size;
  const float* hs   = (const float*)d_in[0];
  const float* enc  = (const float*)d_in[1];
  const float* iph  = (const float*)d_in[2];
  const float* Wq   = (const float*)d_in[3];
  const float* Wk   = (const float*)d_in[4];
  const float* Wv   = (const float*)d_in[5];
  const float* Wkip = (const float*)d_in[6];
  const float* Wvip = (const float*)d_in[7];
  const float* Wo   = (const float*)d_in[8];
  const float* bo   = (const float*)d_in[9];
  float* out = (float*)d_out;

  const int Bb = 8, S = 4096, HID = 1280, CR = 2048, TT = 77, TI = 4;
  size_t off = 0;
  auto alloc = [&](size_t elems) -> u16* {
    u16* p = (u16*)((char*)d_ws + off);
    off += ((elems * 2) + 255) & ~(size_t)255;
    return p;
  };
  u16* hid_bf = alloc((size_t)Bb * S * HID);
  u16* enc_bf = alloc((size_t)Bb * TT * CR);
  u16* ip_bf  = alloc((size_t)Bb * TI * CR);
  u16* WqT  = alloc((size_t)HID * HID);
  u16* WoT  = alloc((size_t)HID * HID);
  u16* WkT  = alloc((size_t)CR * HID);
  u16* WvT  = alloc((size_t)CR * HID);
  u16* WkiT = alloc((size_t)CR * HID);
  u16* WviT = alloc((size_t)CR * HID);
  u16* Qbf  = alloc((size_t)Bb * S * HID);
  u16* Ktx  = alloc((size_t)Bb * TT * HID);
  u16* Vtx  = alloc((size_t)Bb * TT * HID);
  u16* Kib  = alloc((size_t)Bb * TI * HID);
  u16* Vib  = alloc((size_t)Bb * TI * HID);
  u16* attn_bf = hid_bf;   // hid_bf dead after Q projection -> reuse

  auto cvt = [&](const float* src, u16* dst, long long n) {
    cvt_f32_bf16<<<dim3((unsigned)((n + 1023) / 1024)), dim3(256), 0, stream>>>(src, dst, n);
  };
  cvt(hs,  hid_bf, (long long)Bb * S * HID);
  cvt(enc, enc_bf, (long long)Bb * TT * CR);
  cvt(iph, ip_bf,  (long long)Bb * TI * CR);

  transpose_cvt<<<dim3(HID / 32, HID / 32), 256, 0, stream>>>(Wq,   WqT,  HID, HID);
  transpose_cvt<<<dim3(HID / 32, HID / 32), 256, 0, stream>>>(Wo,   WoT,  HID, HID);
  transpose_cvt<<<dim3(HID / 32, CR / 32),  256, 0, stream>>>(Wk,   WkT,  CR,  HID);
  transpose_cvt<<<dim3(HID / 32, CR / 32),  256, 0, stream>>>(Wv,   WvT,  CR,  HID);
  transpose_cvt<<<dim3(HID / 32, CR / 32),  256, 0, stream>>>(Wkip, WkiT, CR,  HID);
  transpose_cvt<<<dim3(HID / 32, CR / 32),  256, 0, stream>>>(Wvip, WviT, CR,  HID);

  // Q = hidden @ Wq ; K/V projections (bf16 outputs)
  gemm_wmma<true, false><<<dim3((Bb * S + 127) / 128, HID / 128), 256, 0, stream>>>(
      hid_bf, WqT, Qbf, nullptr, Bb * S, HID, HID);
  gemm_wmma<true, false><<<dim3((Bb * TT + 127) / 128, HID / 128), 256, 0, stream>>>(
      enc_bf, WkT, Ktx, nullptr, Bb * TT, HID, CR);
  gemm_wmma<true, false><<<dim3((Bb * TT + 127) / 128, HID / 128), 256, 0, stream>>>(
      enc_bf, WvT, Vtx, nullptr, Bb * TT, HID, CR);
  gemm_wmma<true, false><<<dim3((Bb * TI + 127) / 128, HID / 128), 256, 0, stream>>>(
      ip_bf, WkiT, Kib, nullptr, Bb * TI, HID, CR);
  gemm_wmma<true, false><<<dim3((Bb * TI + 127) / 128, HID / 128), 256, 0, stream>>>(
      ip_bf, WviT, Vib, nullptr, Bb * TI, HID, CR);

  // fused softmax-attention (text + ip branches)
  attn_wmma<<<dim3(S / 64, 20, Bb), 128, 0, stream>>>(Qbf, Ktx, Vtx, Kib, Vib, attn_bf);

  // out = attn @ Wo + bo  (fp32 output)
  gemm_wmma<false, true><<<dim3((Bb * S + 127) / 128, HID / 128), 256, 0, stream>>>(
      attn_bf, WoT, out, bo, Bb * S, HID, HID);
}
